// GATModel_25795573580200
// MI455X (gfx1250) — matrix operand order repaired and checked by reading, not compiled, not run on hardware
//
#include <hip/hip_runtime.h>
#include <hip/hip_bf16.h>
#include <stdint.h>
#include <stddef.h>

// ---------------------------------------------------------------------------
// Types for CDNA5 WMMA (wave32)
// ---------------------------------------------------------------------------
typedef __bf16 bf16_t;
typedef bf16_t v16bf __attribute__((ext_vector_type(16)));
typedef float  v8f   __attribute__((ext_vector_type(8)));

#define NEG_SLOPE 0.2f
#define EPS_GAT   1e-16f

// ---------------------------------------------------------------------------
// WMMA fragment packing (per CDNA5 ISA 7.12.2), stored pre-swizzled so each
// lane's 16 bf16 values are 32 contiguous bytes.
//
// A 16x32 (MxK): lane = half*16 + (row&15);
//   elem j<8  -> k = half*8 + j ; elem j>=8 -> k = 16 + half*8 + (j-8)
// B 32x16 (KxN): lane = half*16 + (col&15); same k order as A
//
// Packed index (in bf16 elements): ((tile*nkb + kb)*32 + lane)*16 + j
//   A tile = row/16 over all k-blocks;  B tile = col/16 over all k-blocks
// ---------------------------------------------------------------------------
__device__ __forceinline__ size_t a_pack_idx(int i, int k, int nkb) {
  int rt = i >> 4, r = i & 15;
  int kb = k >> 5, ko = k & 31;
  int half, j;
  if (ko < 16) { half = ko >> 3; j = ko & 7; }
  else         { half = (ko - 16) >> 3; j = 8 + ((ko - 16) & 7); }
  int lane = (half << 4) | r;
  return ((((size_t)rt * nkb + kb) << 5) + lane) * 16 + j;
}

__device__ __forceinline__ size_t b_pack_idx(int k, int n, int nkb) {
  int ct = n >> 4, c = n & 15;
  int kb = k >> 5, ko = k & 31;
  int half, j;
  if (ko < 16) { half = ko >> 3; j = ko & 7; }
  else         { half = (ko - 16) >> 3; j = 8 + ((ko - 16) & 7); }
  int lane = (half << 4) | c;
  return ((((size_t)ct * nkb + kb) << 5) + lane) * 16 + j;
}

// ---------------------------------------------------------------------------
// Ordered float <-> uint key (monotonic), for atomicMax-based segment max.
// ---------------------------------------------------------------------------
__device__ __forceinline__ unsigned f2key(float f) {
  unsigned u = __float_as_uint(f);
  return (u & 0x80000000u) ? ~u : (u | 0x80000000u);
}
__device__ __forceinline__ float key2f(unsigned k) {
  unsigned u = (k & 0x80000000u) ? (k & 0x7FFFFFFFu) : ~k;
  return __uint_as_float(u);
}

// ---------------------------------------------------------------------------
// Elementwise helpers
// ---------------------------------------------------------------------------
__global__ void k_zero32(unsigned* __restrict__ p, long n) {
  long t = (long)blockIdx.x * blockDim.x + threadIdx.x;
  if (t < n) p[t] = 0u;
}

// Pack f32 activation matrix [Nn,K] into A-fragment order (bf16)
__global__ void k_pack_a(const float* __restrict__ X, bf16_t* __restrict__ out,
                         int Nn, int K) {
  long t = (long)blockIdx.x * blockDim.x + threadIdx.x;
  long total = (long)Nn * K;
  if (t >= total) return;
  int i = (int)(t / K);
  int k = (int)(t - (long)i * K);
  out[a_pack_idx(i, k, K >> 5)] = (bf16_t)X[t];
}

// Pack f32 weight matrix [K, Nc] into B-fragment order (bf16), padded to Np cols
__global__ void k_pack_b(const float* __restrict__ W, bf16_t* __restrict__ out,
                         int K, int Nc, int Np) {
  int t = blockIdx.x * blockDim.x + threadIdx.x;
  int total = K * Np;
  if (t >= total) return;
  int k = t / Np;
  int n = t - k * Np;
  float v = (n < Nc) ? W[(size_t)k * Nc + n] : 0.0f;
  out[b_pack_idx(k, n, K >> 5)] = (bf16_t)v;
}

// Next-layer input: pack bf16(relu(agg + b)) directly into A-fragment order
__global__ void k_bias_relu_pack(const float* __restrict__ agg, const float* __restrict__ b,
                                 bf16_t* __restrict__ out, int Nn, int Fo) {
  long t = (long)blockIdx.x * blockDim.x + threadIdx.x;
  long total = (long)Nn * Fo;
  if (t >= total) return;
  int i = (int)(t / Fo);
  int k = (int)(t - (long)i * Fo);
  float v = agg[t] + b[k];
  out[a_pack_idx(i, k, Fo >> 5)] = (bf16_t)(v > 0.f ? v : 0.f);
}

// Build int32 src/dst arrays with self-loops appended.
__global__ void k_build_edges(const long long* __restrict__ ei, int* __restrict__ src,
                              int* __restrict__ dst, int E, int Nn) {
  long t = (long)blockIdx.x * blockDim.x + threadIdx.x;
  long EE = (long)E + Nn;
  if (t >= EE) return;
  if (t < E) {
    src[t] = (int)ei[t];
    dst[t] = (int)ei[(long)E + t];
  } else {
    int i = (int)(t - E);
    src[t] = i;
    dst[t] = i;
  }
}

// ---------------------------------------------------------------------------
// Packed-fragment bf16 GEMM: each wave computes TWO 16x16 output tiles
// (rows rt0, rt0+1 ; one col tile), reusing the B fragment across both.
// Inner loop: 3 x 32B contiguous lane loads + 2 x v_wmma_f32_16x16x32_bf16.
// M % 32 == 0, K % 32 == 0, Np % 16 == 0.
// ---------------------------------------------------------------------------
__global__ void k_gemm_packed_wmma(const bf16_t* __restrict__ A,
                                   const bf16_t* __restrict__ B,
                                   float* __restrict__ C,
                                   int M, int K, int Np) {
  int wid  = (int)(((long)blockIdx.x * blockDim.x + threadIdx.x) >> 5);
  int lane = threadIdx.x & 31;
  int ctiles  = Np >> 4;
  int rgroups = M >> 5;                 // 2 row tiles per wave
  if (wid >= rgroups * ctiles) return;  // uniform per-wave exit
  int rg = wid / ctiles;
  int ct = wid - rg * ctiles;
  int rt0 = rg << 1;
  int nkb = K >> 5;

  const v16bf* Ap0 = (const v16bf*)A + (((size_t)rt0 * nkb) << 5) + lane;
  const v16bf* Ap1 = Ap0 + ((size_t)nkb << 5);
  const v16bf* Bp  = (const v16bf*)B + (((size_t)ct * nkb) << 5) + lane;

  v8f acc0 = {};
  v8f acc1 = {};
  for (int kb = 0; kb < nkb; ++kb) {
    v16bf bfrag = Bp[(size_t)kb << 5];
    v16bf a0    = Ap0[(size_t)kb << 5];
    v16bf a1    = Ap1[(size_t)kb << 5];
    acc0 = __builtin_amdgcn_wmma_f32_16x16x32_bf16(false, a0, false, bfrag,
                                                   (short)0, acc0, false, false);
    acc1 = __builtin_amdgcn_wmma_f32_16x16x32_bf16(false, a1, false, bfrag,
                                                   (short)0, acc1, false, false);
  }

  int half = lane >> 4, l15 = lane & 15;
  int n = (ct << 4) + l15;
  float* C0 = C + (size_t)((rt0 << 4) + (half << 3)) * Np + n;
  float* C1 = C0 + (size_t)16 * Np;
#pragma unroll
  for (int r = 0; r < 8; ++r) C0[(size_t)r * Np] = acc0[r];
#pragma unroll
  for (int r = 0; r < 8; ++r) C1[(size_t)r * Np] = acc1[r];
}

// ---------------------------------------------------------------------------
// Per-node attention coefficients: alpha_s[i] = h_i . a_src ; alpha_d likewise
// ---------------------------------------------------------------------------
__global__ void k_alpha(const float* __restrict__ h, const float* __restrict__ avs,
                        const float* __restrict__ avd, float* __restrict__ als,
                        float* __restrict__ ald, int Nn, int Fo, int stride) {
  int i = blockIdx.x * blockDim.x + threadIdx.x;
  if (i >= Nn) return;
  const float* hp = h + (size_t)i * stride;
  float s = 0.f, d = 0.f;
  for (int k = 0; k < Fo; ++k) {
    float v = hp[k];
    s += v * avs[k];
    d += v * avd[k];
  }
  als[i] = s;
  ald[i] = d;
}

// ---------------------------------------------------------------------------
// Edge passes: logits + segment max, exp + segment sum, normalize, aggregate
// ---------------------------------------------------------------------------
__global__ void k_edge_logit_max(const int* __restrict__ src, const int* __restrict__ dst,
                                 const float* __restrict__ als, const float* __restrict__ ald,
                                 float* __restrict__ logit, unsigned* __restrict__ mkey,
                                 long EE) {
  long e = (long)blockIdx.x * blockDim.x + threadIdx.x;
  if (e >= EE) return;
  float v = als[src[e]] + ald[dst[e]];
  v = (v >= 0.f) ? v : NEG_SLOPE * v;
  logit[e] = v;
  atomicMax(mkey + dst[e], f2key(v));
}

__global__ void k_edge_exp_sum(const int* __restrict__ dst, const float* __restrict__ logit,
                               const unsigned* __restrict__ mkey, float* __restrict__ expw,
                               float* __restrict__ ssum, long EE) {
  long e = (long)blockIdx.x * blockDim.x + threadIdx.x;
  if (e >= EE) return;
  int d = dst[e];
  float w = __expf(logit[e] - key2f(mkey[d]));
  expw[e] = w;
  atomicAdd(ssum + d, w);
}

__global__ void k_edge_coeff(const int* __restrict__ dst, float* __restrict__ expw,
                             const float* __restrict__ ssum, long EE) {
  long e = (long)blockIdx.x * blockDim.x + threadIdx.x;
  if (e >= EE) return;
  expw[e] = expw[e] / (ssum[dst[e]] + EPS_GAT);
}

// One thread per (edge, 4-feature quad): agg[dst] += coeff * h[src]
__global__ void k_edge_agg(const int* __restrict__ src, const int* __restrict__ dst,
                           const float* __restrict__ coeff, const float* __restrict__ h,
                           int stride, int Fo, float* __restrict__ agg, long EE) {
  long t = (long)blockIdx.x * blockDim.x + threadIdx.x;
  int quads = Fo >> 2;
  long total = EE * quads;
  if (t >= total) return;
  long e = t / quads;
  int  q = (int)(t - e * quads);
  int  k = q << 2;
  float c = coeff[e];
  const float4 hv = *(const float4*)(h + (size_t)src[e] * stride + k);
  float* ap = agg + (size_t)dst[e] * Fo + k;
  atomicAdd(ap + 0, c * hv.x);
  atomicAdd(ap + 1, c * hv.y);
  atomicAdd(ap + 2, c * hv.z);
  atomicAdd(ap + 3, c * hv.w);
}

// Final: log_softmax(agg3 + b3) per row of 40 -> d_out
__global__ void k_logsoftmax(const float* __restrict__ agg, const float* __restrict__ b,
                             float* __restrict__ out, int Nn) {
  int i = blockIdx.x * blockDim.x + threadIdx.x;
  if (i >= Nn) return;
  const float* ap = agg + (size_t)i * 40;
  float v[40];
  float mx = -3.4e38f;
#pragma unroll
  for (int c = 0; c < 40; ++c) {
    v[c] = ap[c] + b[c];
    mx = v[c] > mx ? v[c] : mx;
  }
  float s = 0.f;
#pragma unroll
  for (int c = 0; c < 40; ++c) s += __expf(v[c] - mx);
  float lse = mx + __logf(s);
  float* op = out + (size_t)i * 40;
#pragma unroll
  for (int c = 0; c < 40; ++c) op[c] = v[c] - lse;
}

// ---------------------------------------------------------------------------
// Host-side orchestration
// ---------------------------------------------------------------------------
static inline int nblk(long n, int bs) { return (int)((n + bs - 1) / bs); }

extern "C" void kernel_launch(void* const* d_in, const int* in_sizes, int n_in,
                              void* d_out, int out_size, void* d_ws, size_t ws_size,
                              hipStream_t stream) {
  const float*     x   = (const float*)d_in[0];
  const long long* ei  = (const long long*)d_in[1];
  const float* W1  = (const float*)d_in[2];
  const float* as1 = (const float*)d_in[3];
  const float* ad1 = (const float*)d_in[4];
  const float* b1  = (const float*)d_in[5];
  const float* W2  = (const float*)d_in[6];
  const float* as2 = (const float*)d_in[7];
  const float* ad2 = (const float*)d_in[8];
  const float* b2  = (const float*)d_in[9];
  const float* W3  = (const float*)d_in[10];
  const float* as3 = (const float*)d_in[11];
  const float* ad3 = (const float*)d_in[12];
  const float* b3  = (const float*)d_in[13];

  const int FIN = 128, H = 64, C = 40, CP = 48;
  const int Nn = in_sizes[0] / FIN;      // 100000 (divisible by 32)
  const int E  = in_sizes[1] / 2;        // 1600000
  const long EE = (long)E + Nn;          // edges incl. self-loops

  // Carve workspace
  char* w = (char*)d_ws;
  auto carve = [&](size_t bytes) -> char* {
    char* p = w;
    w += (bytes + 255) & ~(size_t)255;
    return p;
  };
  bf16_t*   xb   = (bf16_t*)carve((size_t)Nn * FIN * sizeof(bf16_t));  // packed A, layer1
  bf16_t*   w1b  = (bf16_t*)carve((size_t)FIN * H * sizeof(bf16_t));   // packed B
  bf16_t*   w2b  = (bf16_t*)carve((size_t)H * H * sizeof(bf16_t));
  bf16_t*   w3b  = (bf16_t*)carve((size_t)H * CP * sizeof(bf16_t));
  float*    h1   = (float*)carve((size_t)Nn * H * sizeof(float));
  float*    h2   = (float*)carve((size_t)Nn * H * sizeof(float));
  float*    h3   = (float*)carve((size_t)Nn * CP * sizeof(float));
  bf16_t*   x2b  = (bf16_t*)carve((size_t)Nn * H * sizeof(bf16_t));    // packed A, layer2
  bf16_t*   x3b  = (bf16_t*)carve((size_t)Nn * H * sizeof(bf16_t));    // packed A, layer3
  float*    agg  = (float*)carve((size_t)Nn * H * sizeof(float));      // reused per layer
  float*    als  = (float*)carve((size_t)Nn * sizeof(float));
  float*    ald  = (float*)carve((size_t)Nn * sizeof(float));
  unsigned* mkey = (unsigned*)carve((size_t)Nn * sizeof(unsigned));
  float*    ssum = (float*)carve((size_t)Nn * sizeof(float));
  int*      srcv = (int*)carve((size_t)EE * sizeof(int));
  int*      dstv = (int*)carve((size_t)EE * sizeof(int));
  float*    lgt  = (float*)carve((size_t)EE * sizeof(float));
  float*    expw = (float*)carve((size_t)EE * sizeof(float));
  (void)ws_size; (void)n_in; (void)out_size;

  const int BS = 256;

  // Pre-pack operands + edge arrays
  k_pack_a<<<nblk((long)Nn * FIN, BS), BS, 0, stream>>>(x, xb, Nn, FIN);
  k_pack_b<<<nblk(FIN * H, BS), BS, 0, stream>>>(W1, w1b, FIN, H, H);
  k_pack_b<<<nblk(H * H, BS), BS, 0, stream>>>(W2, w2b, H, H, H);
  k_pack_b<<<nblk(H * CP, BS), BS, 0, stream>>>(W3, w3b, H, C, CP);
  k_build_edges<<<nblk(EE, BS), BS, 0, stream>>>(ei, srcv, dstv, E, Nn);

  // One GAT layer (softmax + aggregate), given precomputed h / alphas
  auto gat_edges = [&](const float* h, int stride, int Fo) {
    k_zero32<<<nblk(Nn, BS), BS, 0, stream>>>(mkey, Nn);
    k_zero32<<<nblk(Nn, BS), BS, 0, stream>>>((unsigned*)ssum, Nn);
    k_zero32<<<nblk((long)Nn * Fo, BS), BS, 0, stream>>>((unsigned*)agg, (long)Nn * Fo);
    k_edge_logit_max<<<nblk(EE, BS), BS, 0, stream>>>(srcv, dstv, als, ald, lgt, mkey, EE);
    k_edge_exp_sum<<<nblk(EE, BS), BS, 0, stream>>>(dstv, lgt, mkey, expw, ssum, EE);
    k_edge_coeff<<<nblk(EE, BS), BS, 0, stream>>>(dstv, expw, ssum, EE);
    k_edge_agg<<<nblk(EE * (Fo >> 2), BS), BS, 0, stream>>>(srcv, dstv, expw, h, stride, Fo,
                                                            agg, EE);
  };

  // GEMM launcher: waves = (M/32)*(Np/16), 8 waves (256 threads) per block
  auto gemm = [&](const bf16_t* A, const bf16_t* B, float* Cout, int K, int Np) {
    long waves = (long)(Nn >> 5) * (Np >> 4);
    k_gemm_packed_wmma<<<nblk(waves * 32, BS), BS, 0, stream>>>(A, B, Cout, Nn, K, Np);
  };

  // ---- Layer 1: FIN -> H
  gemm(xb, w1b, h1, FIN, H);
  k_alpha<<<nblk(Nn, BS), BS, 0, stream>>>(h1, as1, ad1, als, ald, Nn, H, H);
  gat_edges(h1, H, H);
  k_bias_relu_pack<<<nblk((long)Nn * H, BS), BS, 0, stream>>>(agg, b1, x2b, Nn, H);

  // ---- Layer 2: H -> H
  gemm(x2b, w2b, h2, H, H);
  k_alpha<<<nblk(Nn, BS), BS, 0, stream>>>(h2, as2, ad2, als, ald, Nn, H, H);
  gat_edges(h2, H, H);
  k_bias_relu_pack<<<nblk((long)Nn * H, BS), BS, 0, stream>>>(agg, b2, x3b, Nn, H);

  // ---- Layer 3: H -> C (padded to CP for WMMA)
  gemm(x3b, w3b, h3, H, CP);
  k_alpha<<<nblk(Nn, BS), BS, 0, stream>>>(h3, as3, ad3, als, ald, Nn, C, CP);
  gat_edges(h3, CP, C);
  k_logsoftmax<<<nblk(Nn, BS), BS, 0, stream>>>(agg, b3, (float*)d_out, Nn);
}
